// DHSFNN_62079457296496
// MI455X (gfx1250) — matrix operand order repaired and checked by reading, not compiled, not run on hardware
//
#include <hip/hip_runtime.h>
#include <hip/hip_bf16.h>
#include <stdint.h>

// ---------------------------------------------------------------------------
// DH-SFNN on gfx1250 (MI455X).
// Phase 0a: convert W1 (pad K 700->704) and W2 to f16 in workspace.
// Phase 0b: convert x to padded f16 [64000 x 704] (L2-resident A operand).
// Phase 1 : one big WMMA f16 GEMM over all timesteps (k-loop fully unrolled;
//           scheduler hoists loads across WMMAs without copy chains):
//           U1[b*T+t][n] = x_t @ W1^T + b1   (64000 x 1024, f16 out).
// Phase 2 : persistent per-batch-block scan (16 WGs x 512 thr) running the
//           250-step recurrence; layer-2 GEMM via WMMA f16; per-step U1
//           tiles double-buffered into LDS by the Tensor Data Mover.
// ---------------------------------------------------------------------------

typedef __attribute__((ext_vector_type(16))) _Float16     v16h;
typedef __attribute__((ext_vector_type(8)))  _Float16     v8h;
typedef __attribute__((ext_vector_type(8)))  float        v8f;
typedef __attribute__((ext_vector_type(4)))  unsigned int v4u;
typedef __attribute__((ext_vector_type(8)))  int          v8i;
typedef __attribute__((ext_vector_type(4)))  int          v4i;

#define BB   256
#define TT   250
#define INF_ 700
#define KPAD 704
#define HH   256
#define OO   20
#define N1   1024        // H*K branches
#define MTOT (BB*TT)     // 64000
#define G1_BLOCKS ((MTOT/16)*16/8)   // 64000 wave-jobs / 8 waves = 8000
#define KSTEPS (KPAD/32)             // 22

__device__ __forceinline__ float sigm(float x) { return 1.0f / (1.0f + __expf(-x)); }

// -------------------- Phase 0a: weight conversion --------------------------
__global__ void __launch_bounds__(256) prep_weights(
    const float* __restrict__ W1, const float* __restrict__ W2,
    _Float16* __restrict__ W1h, _Float16* __restrict__ W2h)
{
    int n = blockIdx.x; // 0..1023
    for (int k = threadIdx.x; k < KPAD; k += blockDim.x) {
        W1h[(size_t)n * KPAD + k] =
            (k < INF_) ? (_Float16)W1[(size_t)n * INF_ + k] : (_Float16)0.0f;
        if (k < HH)
            W2h[(size_t)n * HH + k] = (_Float16)W2[(size_t)n * HH + k];
    }
}

// -------------------- Phase 0b: input conversion (pad 700 -> 704) ----------
__global__ void __launch_bounds__(256) prep_x(
    const float* __restrict__ x, _Float16* __restrict__ X16)
{
    int row = blockIdx.x;            // 0..63999
    const float* xr = x + (size_t)row * INF_;
    _Float16* o = X16 + (size_t)row * KPAD;
    for (int k = threadIdx.x; k < KPAD; k += 256)
        o[k] = (k < INF_) ? (_Float16)xr[k] : (_Float16)0.0f;
}

// -------------------- fragment loaders -------------------------------------
__device__ __forceinline__ v16h load_a_frag(const _Float16* xrow, int kk, int koff)
{
    const _Float16* ap = xrow + kk + koff;
    v8h a0 = *(const v8h*)ap;
    v8h a1 = *(const v8h*)(ap + 16);
    v16h a;
    #pragma unroll
    for (int i = 0; i < 8; i++) { a[i] = a0[i]; a[i + 8] = a1[i]; }
    return a;
}

__device__ __forceinline__ v16h load_b_frag(const _Float16* wrow, int kk, int khi)
{
    const _Float16* wp = wrow + kk + khi;
    v8h b0v = *(const v8h*)wp;
    v8h b1v = *(const v8h*)(wp + 16);
    v16h bb;
    #pragma unroll
    for (int i = 0; i < 8; i++) { bb[i] = b0v[i]; bb[i + 8] = b1v[i]; }
    return bb;
}

// -------------------- Phase 1: big GEMM1 over all timesteps ----------------
// Each wave computes a 16x64 strip of U1 = X @ W1^T + b1. The 22-step k-loop
// is fully unrolled: fragments are fresh SSA values each step, so the
// scheduler overlaps each step's 10 b128 loads with the previous WMMAs
// without register-copy chains.
// A layout (16x32 f16): lanes 0-15 row=lane,  e0..7 = K kk+0..7,  e8..15 = K kk+16..23
//                       lanes16-31 row=lane-16,e0..7 = K kk+8..15, e8..15 = K kk+24..31
// B layout (32x16 f16): lanes 0-15 col=lane  e0..7 = K kk+0..7, e8..15 = K kk+16..23;
//                       lanes16-31 col=lane-16 e0..7 = K kk+8..15, e8..15 = K kk+24..31.
__global__ void __launch_bounds__(256) gemm1_kernel(
    const _Float16* __restrict__ X16, const _Float16* __restrict__ W1h,
    const float* __restrict__ b1, _Float16* __restrict__ U1)
{
    const int lane = threadIdx.x & 31;
    const int wave = threadIdx.x >> 5;
    const int job  = blockIdx.x * 8 + wave;   // 8000 blocks * 8 waves = 64000 jobs
    const int mTile = job >> 4;               // 0..3999
    const int nBlk  = job & 15;               // 0..15
    const int m0 = mTile << 4;
    const int n0 = nBlk << 6;
    const int row  = m0 + (lane & 15);
    const int koff = (lane < 16) ? 0 : 8;     // A-side sub-offset
    const int khi  = (lane < 16) ? 0 : 8;     // B-side sub-offset
    const int Mb   = (lane < 16) ? 0 : 8;
    const _Float16* xrow = X16 + (size_t)row * KPAD;
    const _Float16* wrow[4];
    #pragma unroll
    for (int j = 0; j < 4; j++)
        wrow[j] = W1h + (size_t)(n0 + j * 16 + (lane & 15)) * KPAD;

    v8f c[4];
    #pragma unroll
    for (int j = 0; j < 4; j++)
        #pragma unroll
        for (int i = 0; i < 8; i++) c[j][i] = 0.0f;

    #pragma unroll
    for (int s = 0; s < KSTEPS; s++) {
        const int kk = s * 32;
        v16h a = load_a_frag(xrow, kk, koff);
        #pragma unroll
        for (int j = 0; j < 4; j++) {
            v16h bbf = load_b_frag(wrow[j], kk, khi);
            c[j] = __builtin_amdgcn_wmma_f32_16x16x32_f16(
                false, a, false, bbf, (short)0, c[j], false, false);
        }
    }
    // epilogue: +bias, store f16 (C layout: lanes 0-15 N=lane M=v; lanes16-31 M=v+8)
    #pragma unroll
    for (int j = 0; j < 4; j++) {
        int n = n0 + j * 16 + (lane & 15);
        float bias = b1[n];
        #pragma unroll
        for (int v = 0; v < 8; v++) {
            int mrow = m0 + v + Mb;
            U1[(size_t)mrow * N1 + n] = (_Float16)(c[j][v] + bias);
        }
    }
}

// -------------------- TDM: 2D tile load U1 -> LDS --------------------------
// Tile: 16 rows (batch) x 1024 f16, row stride TT*N1 elements. Contiguous in LDS.
// D# built per CDNA5 ISA (2D: groups 2/3 zero).
// This toolchain exposes the 6-arg builtin:
//   (uint32x4 g0, int32x8 g1, int32x4 g2, int32x4 g3, int32x8 g4, i32 cpol)
__device__ __forceinline__ void tdm_load_tile(const _Float16* U1, int b0, int t,
                                              unsigned lds_off)
{
    unsigned long long ga =
        (unsigned long long)(uintptr_t)(U1 + ((size_t)b0 * TT + t) * N1);
    v4u g0;
    g0[0] = 1u;                                           // count=1, user mode
    g0[1] = lds_off;                                      // lds_addr (bytes)
    g0[2] = (unsigned)ga;                                 // global_addr[31:0]
    g0[3] = ((unsigned)(ga >> 32) & 0x01FFFFFFu)          // global_addr[56:32]
          | 0x80000000u;                                  // type=2 ("image")
    v8i g1;
    g1[0] = 0x00010000;                                   // data_size=1 (2 bytes)
    g1[1] = (int)(((unsigned)N1 & 0xFFFFu) << 16);        // tensor_dim0[15:0]=1024
    g1[2] = (int)((unsigned)N1 >> 16);                    // dim0 hi16 | dim1 lo16(=0)
    g1[3] = (int)(((1u << 20) >> 16) |                    // tensor_dim1 hi16 (dim1=1M)
                  ((unsigned)N1 << 16));                  // tile_dim0 = 1024
    g1[4] = 16;                                           // tile_dim1=16, tile_dim2=0
    g1[5] = TT * N1;                                      // dim0_stride lo32 = 256000
    g1[6] = 0;                                            // dim0_stride hi16 | dim1_stride lo16
    g1[7] = 0;                                            // dim1_stride hi32
    v4i gz; gz[0] = 0; gz[1] = 0; gz[2] = 0; gz[3] = 0;   // groups 2/3 unused (2D)
    v8i gz8;
    #pragma unroll
    for (int i = 0; i < 8; i++) gz8[i] = 0;               // trailing group: unused
    __builtin_amdgcn_tensor_load_to_lds(g0, g1, gz, gz, gz8, 0);
}

// -------------------- Phase 2: sequential scan -----------------------------
// 16 workgroups x 512 threads (16 waves). WG owns batch rows b0..b0+15 for all
// 250 steps. Wave w is both (a) elementwise owner of batch row r=w and
// (b) GEMM2 worker for N-tiles w*4..w*4+3. Wave 0 drives the TDM pipeline:
// issue tile t+1 at the top of step t, wait TENSORcnt at the bottom so the
// end-of-iteration barrier doubles as the publish barrier (4 barriers/step).
#define UB_OFF   0
#define D2_OFF   65536
#define S2_OFF   (65536 + 65536)
#define BETA1_OFF (S2_OFF + 16384)
#define BETA2_OFF (BETA1_OFF + 4096)
#define B2S_OFF   (BETA2_OFF + 4096)
#define S1H_OFF   (B2S_OFF + 4096)
#define SMEM_TOTAL (S1H_OFF + 8192)      // 167936 bytes

__global__ void __launch_bounds__(512) scan_kernel(
    const _Float16* __restrict__ U1, const _Float16* __restrict__ W2h,
    const float* __restrict__ b2,   const float* __restrict__ tau_n1,
    const float* __restrict__ tau_m1, const float* __restrict__ tau_n2,
    const float* __restrict__ tau_m2, const float* __restrict__ Wr,
    const float* __restrict__ br,   const float* __restrict__ tau_mr,
    const int* __restrict__ warmp,  float* __restrict__ out)
{
    extern __shared__ char smem[];
    _Float16* ubuf  = (_Float16*)(smem + UB_OFF);     // 2 x (16x1024) f16 TDM buffers
    float*    d2    = (float*)(smem + D2_OFF);        // 16*1024 f32
    float*    s2s   = (float*)(smem + S2_OFF);        // 16*256  f32
    float*    beta1 = (float*)(smem + BETA1_OFF);     // 1024 f32
    float*    beta2 = (float*)(smem + BETA2_OFF);     // 1024 f32
    float*    b2s   = (float*)(smem + B2S_OFF);       // 1024 f32
    _Float16* s1h   = (_Float16*)(smem + S1H_OFF);    // 16*256 f16

    const int tid  = threadIdx.x;
    const int lane = tid & 31;
    const int wave = tid >> 5;
    const int b0   = blockIdx.x * 16;
    const int warm = *warmp;

    // --- one-time LDS tables + state init ---
    for (int n = tid; n < N1; n += 512) {
        beta1[n] = sigm(tau_n1[n]);
        beta2[n] = sigm(tau_n2[n]);
        b2s[n]   = b2[n];
    }
    for (int i = tid; i < 16 * N1; i += 512) d2[i] = 0.0f;

    const int r1 = wave;            // batch row owned for elementwise stages
    const int c1 = lane;            // 32-wide chunk of the 1024 branch dim
    const int hbase = c1 * 8;       // 8 hidden units owned
    float d1r[32], m1r[8], s1r[8], m2r[8], s2r[8], a1reg[8], a2reg[8];
    #pragma unroll
    for (int i = 0; i < 32; i++) d1r[i] = 0.0f;
    #pragma unroll
    for (int i = 0; i < 8; i++) {
        m1r[i] = 0.0f; s1r[i] = 0.0f; m2r[i] = 0.0f; s2r[i] = 0.0f;
        a1reg[i] = sigm(tau_m1[hbase + i]);
        a2reg[i] = sigm(tau_m2[hbase + i]);
    }
    // readout ownership (320 threads: 16 rows x 20 outputs)
    const int ro_r = tid / OO;
    const int ro_o = tid - ro_r * OO;
    float mr = 0.0f, accv = 0.0f, alr = 0.0f, brv = 0.0f;
    if (tid < 16 * OO) { alr = sigm(tau_mr[ro_o]); brv = br[ro_o]; }

    const int koff = (lane < 16) ? 0 : 8;
    const int khi  = (lane < 16) ? 0 : 8;
    const int Mb   = (lane < 16) ? 0 : 8;
    const _Float16* arow = s1h + (lane & 15) * HH;

    // TDM: prime the pipeline with tile t=0 into buffer 0 (blocking).
    const unsigned lds_ub = __builtin_amdgcn_groupstaticsize() + UB_OFF;
    if (wave == 0) {
        tdm_load_tile(U1, b0, 0, lds_ub);
        __builtin_amdgcn_s_wait_tensorcnt(0);
    }
    __syncthreads();   // tile 0 + LDS tables visible to all waves

    for (int t = 0; t < TT; t++) {
        // ---- kick off tile t+1 into the other buffer (its last readers
        //      finished before the end-of-iteration barrier of step t-1)
        if (wave == 0 && t + 1 < TT)
            tdm_load_tile(U1, b0, t + 1, lds_ub + ((unsigned)(t + 1) & 1u) * 32768u);

        // ---- stage 1+2: d1 blend (registers) + layer-1 LIF, emit s1 (f16, LDS)
        const _Float16* up = ubuf + (size_t)((t & 1) * (16 * N1) + r1 * N1 + c1 * 32);
        v16h u0  = *(const v16h*)up;
        v16h u1v = *(const v16h*)(up + 16);
        #pragma unroll
        for (int h8 = 0; h8 < 8; h8++) {
            float ssum = 0.0f;
            #pragma unroll
            for (int j = 0; j < 4; j++) {
                int i = h8 * 4 + j;
                int n = c1 * 32 + i;
                float uv  = (i < 16) ? (float)u0[i] : (float)u1v[i - 16];
                float bet = beta1[n];
                d1r[i] = bet * d1r[i] + (1.0f - bet) * uv;
                ssum  += d1r[i];
            }
            float al = a1reg[h8];
            m1r[h8] = m1r[h8] * al + (1.0f - al) * ssum - s1r[h8]; // VTH=1
            float sp = (m1r[h8] - 1.0f > 0.0f) ? 1.0f : 0.0f;
            s1r[h8] = sp;
            s1h[r1 * HH + hbase + h8] = (_Float16)sp;
        }
        __syncthreads();

        // ---- stage 3: GEMM2 (s1 @ W2^T) via WMMA + d2 blend in WMMA layout
        #pragma unroll
        for (int jt = 0; jt < 4; jt++) {
            int n0 = (wave * 4 + jt) * 16;
            int n  = n0 + (lane & 15);
            v8f c;
            #pragma unroll
            for (int i = 0; i < 8; i++) c[i] = 0.0f;
            #pragma unroll
            for (int kk = 0; kk < HH; kk += 32) {
                v8h a0 = *(const v8h*)(arow + kk + koff);
                v8h a1 = *(const v8h*)(arow + kk + koff + 16);
                v16h a;
                #pragma unroll
                for (int i = 0; i < 8; i++) { a[i] = a0[i]; a[i + 8] = a1[i]; }
                v16h bbf = load_b_frag(W2h + (size_t)n * HH, kk, khi);
                c = __builtin_amdgcn_wmma_f32_16x16x32_f16(
                    false, a, false, bbf, (short)0, c, false, false);
            }
            float bet = beta2[n], bias = b2s[n];
            #pragma unroll
            for (int v = 0; v < 8; v++) {
                int mrow = v + Mb;
                float dv = d2[mrow * N1 + n];
                d2[mrow * N1 + n] = bet * dv + (1.0f - bet) * (c[v] + bias);
            }
        }
        __syncthreads();

        // ---- stage 4: layer-2 LIF from d2 (LDS), emit s2 (f32, LDS)
        #pragma unroll
        for (int h8 = 0; h8 < 8; h8++) {
            float ssum = 0.0f;
            #pragma unroll
            for (int j = 0; j < 4; j++)
                ssum += d2[r1 * N1 + c1 * 32 + h8 * 4 + j];
            float al = a2reg[h8];
            m2r[h8] = m2r[h8] * al + (1.0f - al) * ssum - s2r[h8];
            float sp = (m2r[h8] - 1.0f > 0.0f) ? 1.0f : 0.0f;
            s2r[h8] = sp;
            s2s[r1 * HH + hbase + h8] = sp;
        }
        __syncthreads();

        // ---- stage 5: leaky readout integrator (320 scalar threads)
        if (tid < 16 * OO) {
            const float* wr = Wr + ro_o * HH;
            const float* sv = s2s + ro_r * HH;
            float dot = brv;
            #pragma unroll 8
            for (int h = 0; h < HH; h++) dot += sv[h] * wr[h];
            mr = mr * alr + (1.0f - alr) * dot;
            if (t >= warm) accv += mr;
        }

        // ---- bottom of step: tile t+1 must have landed before next stage 1;
        //      the barrier also publishes it and closes the iteration.
        if (wave == 0 && t + 1 < TT) __builtin_amdgcn_s_wait_tensorcnt(0);
        __syncthreads();
    }

    if (tid < 16 * OO)
        out[(b0 + ro_r) * OO + ro_o] = accv * (1.0f / (float)(TT - warm));
}

// ---------------------------------------------------------------------------
extern "C" void kernel_launch(void* const* d_in, const int* in_sizes, int n_in,
                              void* d_out, int out_size, void* d_ws, size_t ws_size,
                              hipStream_t stream) {
    (void)in_sizes; (void)n_in; (void)out_size; (void)ws_size;
    const float* x      = (const float*)d_in[0];
    const float* W1     = (const float*)d_in[1];
    const float* b1     = (const float*)d_in[2];
    const float* tau_n1 = (const float*)d_in[3];
    const float* tau_m1 = (const float*)d_in[4];
    const float* W2     = (const float*)d_in[5];
    const float* b2     = (const float*)d_in[6];
    const float* tau_n2 = (const float*)d_in[7];
    const float* tau_m2 = (const float*)d_in[8];
    const float* Wr     = (const float*)d_in[9];
    const float* br     = (const float*)d_in[10];
    const float* tau_mr = (const float*)d_in[11];
    const int*   warmp  = (const int*)d_in[12];
    float*       out    = (float*)d_out;

    char* ws = (char*)d_ws;
    _Float16* U1  = (_Float16*)ws;                         // 64000*1024 f16 = 131072000 B
    _Float16* X16 = (_Float16*)(ws + (size_t)MTOT * N1 * 2);           // 64000*704 f16
    _Float16* W1h = (_Float16*)(ws + (size_t)MTOT * N1 * 2
                                   + (size_t)MTOT * KPAD * 2);         // 1024*704 f16
    _Float16* W2h = (_Float16*)(ws + (size_t)MTOT * N1 * 2
                                   + (size_t)MTOT * KPAD * 2
                                   + (size_t)N1 * KPAD * 2);           // 1024*256 f16

    prep_weights<<<N1, 256, 0, stream>>>(W1, W2, W1h, W2h);
    prep_x<<<MTOT, 256, 0, stream>>>(x, X16);
    gemm1_kernel<<<G1_BLOCKS, 256, 0, stream>>>(X16, W1h, b1, U1);
    scan_kernel<<<BB / 16, 512, SMEM_TOTAL, stream>>>(U1, W2h, b2, tau_n1, tau_m1,
                                                      tau_n2, tau_m2, Wr, br, tau_mr,
                                                      warmp, out);
}
